// RNNres_30537217474611
// MI455X (gfx1250) — compile-verified, matrix operand order checked by reading, not run on hardware
//
#include <hip/hip_runtime.h>
#include <cstddef>

#define TT 2048
#define BB 64
#define NH 256
#define NG 1024   // 4*NH
#define KK 512    // NINP + NH
#define FC1N 85   // 256/3
#define RESN 10

typedef __attribute__((ext_vector_type(16))) __bf16 v16bf;
typedef __attribute__((ext_vector_type(8)))  float  v8f;

// ---------- helpers ----------
__device__ __forceinline__ unsigned short f2bf(float x) {
    unsigned u = __float_as_uint(x);
    unsigned r = ((u >> 16) & 1u) + 0x7FFFu;      // round-to-nearest-even
    return (unsigned short)((u + r) >> 16);
}
__device__ __forceinline__ float sigf(float x) { return 1.0f / (1.0f + __expf(-x)); }
__device__ __forceinline__ float tanhfast(float x) {
    float e = __expf(-2.0f * fabsf(x));
    float r = (1.0f - e) / (1.0f + e);
    return copysignf(r, x);
}
__device__ __forceinline__ v16bf ldbf16x16(const unsigned short* p) {
    return *reinterpret_cast<const v16bf*>(p);
}

// ---------- kernel 1: pack weights col-major bf16 + fused bias ----------
// Wc[n*512 + k] = k<256 ? W_ih[n][k] : W_hh[n][k-256]   (column-major per gate col n)
__global__ void convert_kernel(const float* __restrict__ W_ih, const float* __restrict__ W_hh,
                               const float* __restrict__ b_ih, const float* __restrict__ b_hh,
                               unsigned short* __restrict__ Wc, float* __restrict__ biasc) {
    int idx = blockIdx.x * blockDim.x + threadIdx.x;
    if (idx >= NG * KK) return;
    int n = idx >> 9;          // gate column 0..1023
    int k = idx & (KK - 1);    // 0..511
    float v = (k < NH) ? W_ih[n * NH + k] : W_hh[n * NH + (k - NH)];
    Wc[(size_t)n * KK + k] = f2bf(v);
    if (idx < NG) biasc[idx] = b_ih[idx] + b_hh[idx];
}

// ---------- kernel 2: embedding gather -> bf16 [t][b][256] ----------
__global__ void embed_kernel(const int* __restrict__ tokens, const float* __restrict__ emb_w,
                             unsigned short* __restrict__ emb_bf) {
    int idx = blockIdx.x * blockDim.x + threadIdx.x;   // one thread per 8 elems
    if (idx >= TT * BB * (NH / 8)) return;
    int tb = idx >> 5;                 // (t*64 + b)
    int k8 = (idx & 31) * 8;
    int t = tb >> 6, b = tb & 63;
    int tok = tokens[b * TT + t];      // input is [B,T]
    const float* src = emb_w + (size_t)tok * NH + k8;
    alignas(16) unsigned short tmp[8];
#pragma unroll
    for (int j = 0; j < 8; ++j) tmp[j] = f2bf(src[j]);
    *reinterpret_cast<uint4*>(emb_bf + (size_t)tb * NH + k8) = *reinterpret_cast<const uint4*>(tmp);
}

// ---------- kernel 3: persistent LSTM scan (WMMA bf16 -> f32) ----------
// grid = 4 blocks (16 batch rows each), block = 512 threads = 16 waves.
// Wave w owns hidden units u in [w*16, w*16+16): computes the i,f,g,o 16x16 tiles
// for those units -> full cell update stays in-lane.
__global__ void __launch_bounds__(512)
lstm_scan_kernel(const unsigned short* __restrict__ emb_bf, const unsigned short* __restrict__ Wc,
                 const float* __restrict__ biasc, const float* __restrict__ hx,
                 const float* __restrict__ cx, const int* __restrict__ seq,
                 float* __restrict__ out, float* __restrict__ feature) {
    const int tid  = threadIdx.x;
    const int wave = tid >> 5;
    const int lane = tid & 31;
    const int l16  = lane & 15;
    const int hi   = lane >> 4;               // K/M half select
    const int b0   = blockIdx.x * 16;         // batch base for this WG
    const int u    = wave * 16 + l16;         // hidden unit column (per lane)

    __shared__ __align__(64) unsigned short hbf[16 * NH];   // h_prev in bf16, rows = local batch

    // init h from hx
    for (int it = 0; it < 8; ++it) {
        int idx = it * 512 + tid;             // 4096 cells
        int b = idx >> 8, uu = idx & (NH - 1);
        hbf[idx] = f2bf(hx[(size_t)(b0 + b) * NH + uu]);
    }
    __syncthreads();

    // per-lane persistent state: 8 cells = rows (r + 8*hi), column u
    v8f c_reg, pool = {0.f,0.f,0.f,0.f,0.f,0.f,0.f,0.f};
    int iseq[8];
#pragma unroll
    for (int r = 0; r < 8; ++r) {
        int brow = b0 + r + 8 * hi;
        c_reg[r] = cx[(size_t)brow * NH + u];
        iseq[r]  = seq[brow];
    }
    const float bi = biasc[u], bf_ = biasc[NH + u], bg = biasc[2 * NH + u], bo = biasc[3 * NH + u];

    const unsigned short* wbase = Wc + (size_t)u * KK + hi * 16;  // col u, K-half select
    const size_t GSTRIDE = (size_t)NH * KK;                       // col stride between gates

    for (int t = 0; t < TT; ++t) {
        v8f ai = {0.f,0.f,0.f,0.f,0.f,0.f,0.f,0.f};
        v8f af = ai, ag = ai, ao = ai;

        // Opaque ZERO OFFSET (not an opaque pointer): defeats LICM of the
        // loop-invariant weight loads (which would spill ~512 VGPRs to
        // scratch), while preserving pointer provenance so address-space
        // inference still lowers these to global_load_b128 (LOADcnt only),
        // not flat_load_b128 (LOADcnt + DScnt).
        unsigned zero = 0;
        asm volatile("" : "+v"(zero));
        const unsigned short* wb = wbase + zero;

        const unsigned short* embrow = emb_bf + ((size_t)t * BB + b0 + l16) * NH + hi * 16;
        __builtin_prefetch((const void*)(embrow + BB * NH), 0, 1);   // next timestep rows

        // K-tiles 0..7: A = embedding (global)
#pragma unroll
        for (int kt = 0; kt < 8; ++kt) {
            v16bf a = ldbf16x16(embrow + kt * 32);
            const unsigned short* wp = wb + kt * 32;
            ai = __builtin_amdgcn_wmma_f32_16x16x32_bf16(false, a, false, ldbf16x16(wp),               (short)0, ai, false, false);
            af = __builtin_amdgcn_wmma_f32_16x16x32_bf16(false, a, false, ldbf16x16(wp + GSTRIDE),     (short)0, af, false, false);
            ag = __builtin_amdgcn_wmma_f32_16x16x32_bf16(false, a, false, ldbf16x16(wp + 2 * GSTRIDE), (short)0, ag, false, false);
            ao = __builtin_amdgcn_wmma_f32_16x16x32_bf16(false, a, false, ldbf16x16(wp + 3 * GSTRIDE), (short)0, ao, false, false);
        }
        // K-tiles 8..15: A = h_prev (LDS)
#pragma unroll
        for (int kt = 0; kt < 8; ++kt) {
            v16bf a = ldbf16x16(&hbf[l16 * NH + kt * 32 + hi * 16]);
            const unsigned short* wp = wb + NH + kt * 32;
            ai = __builtin_amdgcn_wmma_f32_16x16x32_bf16(false, a, false, ldbf16x16(wp),               (short)0, ai, false, false);
            af = __builtin_amdgcn_wmma_f32_16x16x32_bf16(false, a, false, ldbf16x16(wp + GSTRIDE),     (short)0, af, false, false);
            ag = __builtin_amdgcn_wmma_f32_16x16x32_bf16(false, a, false, ldbf16x16(wp + 2 * GSTRIDE), (short)0, ag, false, false);
            ao = __builtin_amdgcn_wmma_f32_16x16x32_bf16(false, a, false, ldbf16x16(wp + 3 * GSTRIDE), (short)0, ao, false, false);
        }
        __syncthreads();   // all waves done reading hbf

        // in-register cell update for 8 (row, u) cells
#pragma unroll
        for (int r = 0; r < 8; ++r) {
            float gi = ai[r] + bi, gf = af[r] + bf_, gg = ag[r] + bg, go = ao[r] + bo;
            float cn = sigf(gf) * c_reg[r] + sigf(gi) * tanhfast(gg);
            float h  = sigf(go) * tanhfast(cn);
            c_reg[r] = cn;
            int brl = r + 8 * hi;                       // local batch row
            hbf[brl * NH + u] = f2bf(h);
            if (t < iseq[r]) pool[r] += h;
            if (t == TT - 1)                            // final hidden state -> hx_f
                out[2 * BB + (size_t)(b0 + brl) * NH + u] = h;
        }
        __syncthreads();   // hbf published for next step
    }

    // outputs: cx_f and pooled feature
#pragma unroll
    for (int r = 0; r < 8; ++r) {
        int brow = b0 + r + 8 * hi;
        out[2 * BB + BB * NH + (size_t)brow * NH + u] = c_reg[r];
        feature[(size_t)brow * NH + u] = pool[r] / (float)iseq[r];
    }
}

// ---------- kernel 4: ResNet block + classifier head + log-softmax ----------
__global__ void head_kernel(const float* __restrict__ feature,
                            const float* __restrict__ rfc1_w, const float* __restrict__ rfc1_b,
                            const float* __restrict__ rbn_g,  const float* __restrict__ rbn_b,
                            const float* __restrict__ rbn_rm, const float* __restrict__ rbn_rv,
                            const float* __restrict__ rfc2_w, const float* __restrict__ rfc2_b,
                            const float* __restrict__ fc1_w,  const float* __restrict__ fc1_b,
                            const float* __restrict__ bn1_g,  const float* __restrict__ bn1_b,
                            const float* __restrict__ bn1_rm, const float* __restrict__ bn1_rv,
                            const float* __restrict__ fc2_w,  const float* __restrict__ fc2_b,
                            float* __restrict__ out) {
    const int b = blockIdx.x, tid = threadIdx.x;   // 64 blocks x 128 threads
    __shared__ float f[NH], f2[NH], resv[RESN], fcv[FC1N], lg[2];

    f[tid] = feature[(size_t)b * NH + tid];
    f[tid + 128] = feature[(size_t)b * NH + tid + 128];
    __syncthreads();

    if (tid < RESN) {
        float s = rfc1_b[tid];
        for (int k = 0; k < NH; ++k) s += fmaxf(f[k], 0.f) * rfc1_w[tid * NH + k];
        s = fmaxf(s, 0.f);
        s = (s - rbn_rm[tid]) * rsqrtf(rbn_rv[tid] + 1e-5f) * rbn_g[tid] + rbn_b[tid];
        resv[tid] = s;
    }
    __syncthreads();

    for (int j = 0; j < 2; ++j) {
        int n = tid + j * 128;
        float s = f[n] + rfc2_b[n];
        for (int r = 0; r < RESN; ++r) s += resv[r] * rfc2_w[n * RESN + r];
        f2[n] = s;
    }
    __syncthreads();

    if (tid < FC1N) {
        float s = fc1_b[tid];
        for (int k = 0; k < NH; ++k) s += f2[k] * fc1_w[tid * NH + k];
        s = (s - bn1_rm[tid]) * rsqrtf(bn1_rv[tid] + 1e-5f) * bn1_g[tid] + bn1_b[tid];
        s = fmaxf(s, 0.01f * s);                 // leaky relu
        fcv[tid] = s;
    }
    __syncthreads();

    if (tid < 2) {
        float s = fc2_b[tid];
        for (int k = 0; k < FC1N; ++k) s += fcv[k] * fc2_w[tid * FC1N + k];
        lg[tid] = s;
    }
    __syncthreads();

    if (tid == 0) {
        float m = fmaxf(lg[0], lg[1]);
        float lse = m + logf(__expf(lg[0] - m) + __expf(lg[1] - m));
        out[b * 2 + 0] = lg[0] - lse;
        out[b * 2 + 1] = lg[1] - lse;
    }
}

// ---------- launch ----------
extern "C" void kernel_launch(void* const* d_in, const int* in_sizes, int n_in,
                              void* d_out, int out_size, void* d_ws, size_t ws_size,
                              hipStream_t stream) {
    const int*   tokens = (const int*)  d_in[0];
    const float* hx     = (const float*)d_in[1];
    const float* cx     = (const float*)d_in[2];
    const int*   seq    = (const int*)  d_in[3];
    const float* emb_w  = (const float*)d_in[4];
    const float* W_ih   = (const float*)d_in[5];
    const float* W_hh   = (const float*)d_in[6];
    const float* b_ih   = (const float*)d_in[7];
    const float* b_hh   = (const float*)d_in[8];
    const float* rfc1_w = (const float*)d_in[9];
    const float* rfc1_b = (const float*)d_in[10];
    const float* rbn_g  = (const float*)d_in[11];
    const float* rbn_b  = (const float*)d_in[12];
    const float* rbn_rm = (const float*)d_in[13];
    const float* rbn_rv = (const float*)d_in[14];
    const float* rfc2_w = (const float*)d_in[15];
    const float* rfc2_b = (const float*)d_in[16];
    const float* fc1_w  = (const float*)d_in[17];
    const float* fc1_b  = (const float*)d_in[18];
    const float* bn1_g  = (const float*)d_in[19];
    const float* bn1_b  = (const float*)d_in[20];
    const float* bn1_rm = (const float*)d_in[21];
    const float* bn1_rv = (const float*)d_in[22];
    const float* fc2_w  = (const float*)d_in[23];
    const float* fc2_b  = (const float*)d_in[24];
    float* out = (float*)d_out;

    // workspace layout (256B aligned chunks)
    char* ws = (char*)d_ws;
    unsigned short* Wc      = (unsigned short*)(ws);                        // 1 MiB
    float*          biasc   = (float*)(ws + 1048576);                       // 4 KiB
    float*          feat    = (float*)(ws + 1048576 + 4096);                // 64 KiB
    unsigned short* emb_bf  = (unsigned short*)(ws + 1048576 + 4096 + 65536); // 64 MiB

    convert_kernel<<<(NG * KK + 255) / 256, 256, 0, stream>>>(W_ih, W_hh, b_ih, b_hh, Wc, biasc);
    embed_kernel<<<(TT * BB * (NH / 8) + 255) / 256, 256, 0, stream>>>(tokens, emb_w, emb_bf);
    lstm_scan_kernel<<<4, 512, 0, stream>>>(emb_bf, Wc, biasc, hx, cx, seq, out, feat);
    head_kernel<<<BB, 128, 0, stream>>>(feat, rfc1_w, rfc1_b, rbn_g, rbn_b, rbn_rm, rbn_rv,
                                        rfc2_w, rfc2_b, fc1_w, fc1_b, bn1_g, bn1_b, bn1_rm,
                                        bn1_rv, fc2_w, fc2_b, out);
}
